// SerialIntervalGraph_32830730010793
// MI455X (gfx1250) — compile-verified, matrix operand order checked by reading, not run on hardware
//
#include <hip/hip_runtime.h>
#include <hip/hip_bf16.h>

#define BB 16
#define TT 28
#define NN 2048
#define HH 32
#define LLAG 8
#define LN_EPS 1e-5f

typedef __attribute__((ext_vector_type(16))) __bf16 v16bf;
typedef __attribute__((ext_vector_type(8)))  __bf16 v8bf;
typedef __attribute__((ext_vector_type(8)))  float  v8f;
typedef __attribute__((ext_vector_type(4)))  float  f4;

// Load 16 bf16 elements for a WMMA A/B operand row (CDNA5 16-bit operand
// layout): elements e<8 from ptr + half*8, e>=8 from ptr + 16 + half*8.
__device__ __forceinline__ v16bf load_row16(const __bf16* p, int half) {
    v8bf lo = *(const v8bf*)(p + half * 8);
    v8bf hi = *(const v8bf*)(p + 16 + half * 8);
    v16bf r;
#pragma unroll
    for (int i = 0; i < 8; ++i) { r[i] = lo[i]; r[8 + i] = hi[i]; }
    return r;
}

// Same operand layout, but sourced from an fp32 row (converted to bf16).
__device__ __forceinline__ v16bf cvt_row16(const float* p, int half) {
    float lo[8], hi[8];
    *(f4*)&lo[0] = *(const f4*)(p + 8 * half);
    *(f4*)&lo[4] = *(const f4*)(p + 8 * half + 4);
    *(f4*)&hi[0] = *(const f4*)(p + 16 + 8 * half);
    *(f4*)&hi[4] = *(const f4*)(p + 16 + 8 * half + 4);
    v16bf r;
#pragma unroll
    for (int i = 0; i < 8; ++i) { r[i] = (__bf16)lo[i]; r[8 + i] = (__bf16)hi[i]; }
    return r;
}

__device__ __forceinline__ v8bf pack8(v8f a) {
    v8bf r;
#pragma unroll
    for (int i = 0; i < 8; ++i) r[i] = (__bf16)a[i];
    return r;
}

// --- Kernel 0: delay softmax + delayed signal, packed into Vd aux operand ---
// Vd is [B, 16, N] bf16: row 0 = delayed signal, row 1 = 1.0 (denominator
// accumulator row), rows 2..15 uninitialized (their WMMA D rows are never read).
__global__ void k_delay_vd(const float* __restrict__ x, const float* __restrict__ dlog,
                           __bf16* __restrict__ Vd) {
    int tid = blockIdx.x * blockDim.x + threadIdx.x;
    if (tid >= BB * NN) return;
    int b = tid / NN, n = tid % NN;
    float w[LLAG]; float m = -1e30f;
#pragma unroll
    for (int t = 0; t < LLAG; ++t) { w[t] = dlog[t]; m = fmaxf(m, w[t]); }
    float s = 0.f;
#pragma unroll
    for (int t = 0; t < LLAG; ++t) { w[t] = __expf(w[t] - m); s += w[t]; }
    float inv = 1.f / s, acc = 0.f;
#pragma unroll
    for (int t = 0; t < LLAG; ++t)
        acc += w[t] * inv * x[(size_t)b * TT * NN + (size_t)(TT - 1 - t) * NN + n];
    Vd[((size_t)b * 16 + 0) * NN + n] = (__bf16)acc;
    Vd[((size_t)b * 16 + 1) * NN + n] = (__bf16)1.0f;
}

// --- Kernel 1: WMMA Q/K/V projections, one wave per 16-row tile ---
// Q,K computed transposed (A = weight rows -> D rows are output channels,
// lane holds 8 consecutive h) so stores are contiguous v8bf. V computed
// straight (D rows = n) so Vt[B,H,N] stores are contiguous v8bf too.
// Q is pre-scaled by (1-geo)/sqrt(H) so the flash score bias is a single FMA.
__global__ void __launch_bounds__(32) k_qkvw(
    const float* __restrict__ feat,
    const float* __restrict__ Wq, const float* __restrict__ bq,
    const float* __restrict__ Wk, const float* __restrict__ bk,
    const float* __restrict__ Wv, const float* __restrict__ bv,
    const float* __restrict__ geo_w,
    __bf16* __restrict__ Qb, __bf16* __restrict__ Kb, __bf16* __restrict__ Vt) {
    const int b  = blockIdx.x / (NN / 16);
    const int n0 = (blockIdx.x % (NN / 16)) * 16;
    const int lane = threadIdx.x, lm = lane & 15, half = lane >> 4;

    const float geo    = 1.f / (1.f + __expf(-geo_w[0]));
    const float sscale = (1.f - geo) * 0.17677669529663687f;  // (1-geo)/sqrt(32)

    const v16bf bf   = cvt_row16(feat + ((size_t)b * NN + n0 + lm) * HH, half);
    const v16bf awq0 = cvt_row16(Wq + (size_t)lm * HH, half);
    const v16bf awq1 = cvt_row16(Wq + (size_t)(16 + lm) * HH, half);
    const v16bf awk0 = cvt_row16(Wk + (size_t)lm * HH, half);
    const v16bf awk1 = cvt_row16(Wk + (size_t)(16 + lm) * HH, half);
    const v16bf awv0 = cvt_row16(Wv + (size_t)lm * HH, half);
    const v16bf awv1 = cvt_row16(Wv + (size_t)(16 + lm) * HH, half);

    v8f z = {};
    v8f q0 = __builtin_amdgcn_wmma_f32_16x16x32_bf16(false, awq0, false, bf, (short)0, z, false, false);
    v8f q1 = __builtin_amdgcn_wmma_f32_16x16x32_bf16(false, awq1, false, bf, (short)0, z, false, false);
    v8f k0 = __builtin_amdgcn_wmma_f32_16x16x32_bf16(false, awk0, false, bf, (short)0, z, false, false);
    v8f k1 = __builtin_amdgcn_wmma_f32_16x16x32_bf16(false, awk1, false, bf, (short)0, z, false, false);
    v8f v0 = __builtin_amdgcn_wmma_f32_16x16x32_bf16(false, bf, false, awv0, (short)0, z, false, false);
    v8f v1 = __builtin_amdgcn_wmma_f32_16x16x32_bf16(false, bf, false, awv1, (short)0, z, false, false);

    // biases: transposed D -> h = v + 8*half (+16 for tile 1)
    float bq0[8], bq1[8], bk0[8], bk1[8];
    *(f4*)&bq0[0] = *(const f4*)(bq + 8 * half);      *(f4*)&bq0[4] = *(const f4*)(bq + 8 * half + 4);
    *(f4*)&bq1[0] = *(const f4*)(bq + 16 + 8 * half); *(f4*)&bq1[4] = *(const f4*)(bq + 16 + 8 * half + 4);
    *(f4*)&bk0[0] = *(const f4*)(bk + 8 * half);      *(f4*)&bk0[4] = *(const f4*)(bk + 8 * half + 4);
    *(f4*)&bk1[0] = *(const f4*)(bk + 16 + 8 * half); *(f4*)&bk1[4] = *(const f4*)(bk + 16 + 8 * half + 4);
    const float bv0 = bv[lm], bv1 = bv[16 + lm];
#pragma unroll
    for (int v = 0; v < 8; ++v) {
        q0[v] = (q0[v] + bq0[v]) * sscale;
        q1[v] = (q1[v] + bq1[v]) * sscale;
        k0[v] = k0[v] + bk0[v];
        k1[v] = k1[v] + bk1[v];
        v0[v] = v0[v] + bv0;
        v1[v] = v1[v] + bv1;
    }
    // Q/K rows: n = n0+lm (lane-fixed), 8 consecutive h per store
    __bf16* qrow = Qb + ((size_t)b * NN + n0 + lm) * HH;
    __bf16* krow = Kb + ((size_t)b * NN + n0 + lm) * HH;
    *(v8bf*)(qrow + 8 * half)      = pack8(q0);
    *(v8bf*)(qrow + 16 + 8 * half) = pack8(q1);
    *(v8bf*)(krow + 8 * half)      = pack8(k0);
    *(v8bf*)(krow + 16 + 8 * half) = pack8(k1);
    // Vt rows: h = lm (lane-fixed), 8 consecutive n per store
    *(v8bf*)(Vt + ((size_t)b * HH + lm) * NN + n0 + 8 * half)      = pack8(v0);
    *(v8bf*)(Vt + ((size_t)b * HH + 16 + lm) * NN + n0 + 8 * half) = pack8(v1);
}

// --- Kernel 2: flash attention, one wave per (b, 16-row n-tile) ---
__global__ void __launch_bounds__(32) k_flash(
    const __bf16* __restrict__ Qb, const __bf16* __restrict__ Kb,
    const __bf16* __restrict__ Vt, const __bf16* __restrict__ Vd,
    const float* __restrict__ adj, const float* __restrict__ geo_w,
    float* __restrict__ att, float* __restrict__ prop) {
    const int b     = blockIdx.x / (NN / 16);
    const int ntile = (blockIdx.x % (NN / 16)) * 16;
    const int lane  = threadIdx.x;
    const int lm    = lane & 15, half = lane >> 4;
    const int n     = ntile + lm;

    const float geo    = 1.f / (1.f + __expf(-geo_w[0]));
    const float bscale = geo * 5.f;

    const v16bf aq = load_row16(Qb + ((size_t)b * NN + n) * HH, half);

    const float*  adjrow = adj + (size_t)n * NN;
    const __bf16* vrow0  = Vt + ((size_t)b * HH + lm) * NN;
    const __bf16* vrow1  = vrow0 + (size_t)16 * NN;
    const __bf16* drow   = Vd + ((size_t)b * 16 + lm) * NN;
    const __bf16* krow   = Kb + (size_t)b * NN * HH;

    v8f acc0 = {}, acc1 = {}, accp = {};
    float run_max = -1e30f;

    for (int m0 = 0; m0 < NN; m0 += 32) {
        v16bf ak0 = load_row16(krow + (size_t)(m0 + lm) * HH, half);
        v16bf ak1 = load_row16(krow + (size_t)(m0 + 16 + lm) * HH, half);
        // prefetch next iteration's lines (speculative, harmless at tail)
        __builtin_prefetch(krow + (size_t)(m0 + 32 + lm) * HH, 0, 1);
        __builtin_prefetch(krow + (size_t)(m0 + 48 + lm) * HH, 0, 1);
        __builtin_prefetch(adjrow + m0 + 32, 0, 1);

        v8f z = {};
        v8f s0 = __builtin_amdgcn_wmma_f32_16x16x32_bf16(false, ak0, false, aq,
                                                         (short)0, z, false, false);
        v8f s1 = __builtin_amdgcn_wmma_f32_16x16x32_bf16(false, ak1, false, aq,
                                                         (short)0, z, false, false);
        // adj_prior bias: row n, m = m0 + chunk*16 + 8*half + v (contiguous)
        float aj0[8], aj1[8];
        *(f4*)&aj0[0] = *(const f4*)(adjrow + m0 + 8 * half);
        *(f4*)&aj0[4] = *(const f4*)(adjrow + m0 + 8 * half + 4);
        *(f4*)&aj1[0] = *(const f4*)(adjrow + m0 + 16 + 8 * half);
        *(f4*)&aj1[4] = *(const f4*)(adjrow + m0 + 16 + 8 * half + 4);
        float cmax = -1e30f;
#pragma unroll
        for (int v = 0; v < 8; ++v) {
            s0[v] = s0[v] + bscale * aj0[v];   // Q pre-scaled by (1-geo)/sqrt(H)
            s1[v] = s1[v] + bscale * aj1[v];
            cmax = fmaxf(cmax, fmaxf(s0[v], s1[v]));
        }
        cmax = fmaxf(cmax, __shfl_xor(cmax, 16, 32));
        float nmax  = fmaxf(run_max, cmax);
        float alpha = __expf(run_max - nmax);
        run_max = nmax;
        acc0 *= alpha; acc1 *= alpha;
        accp[0] *= alpha; accp[1] *= alpha;  // only rows 0 (prop) and 1 (sum) are live

        v16bf pb;
#pragma unroll
        for (int v = 0; v < 8; ++v) {
            pb[v]     = (__bf16)__expf(s0[v] - nmax);  // S^T D-layout == attn^T B-layout
            pb[v + 8] = (__bf16)__expf(s1[v] - nmax);
        }
        v16bf av0 = load_row16(vrow0 + m0, half);
        v16bf av1 = load_row16(vrow1 + m0, half);
        v16bf avd = load_row16(drow + m0, half);
        acc0 = __builtin_amdgcn_wmma_f32_16x16x32_bf16(false, av0, false, pb,
                                                       (short)0, acc0, false, false);
        acc1 = __builtin_amdgcn_wmma_f32_16x16x32_bf16(false, av1, false, pb,
                                                       (short)0, acc1, false, false);
        accp = __builtin_amdgcn_wmma_f32_16x16x32_bf16(false, avd, false, pb,
                                                       (short)0, accp, false, false);
    }
    // accp row 0 (lanes 0-15, elem 0) = propagated numerator; row 1 (elem 1) = denom
    float tot = __shfl(accp[1], lm, 32);     // broadcast denom to both half-lanes
    float inv = 1.f / tot;
    float* arow = att + ((size_t)b * NN + n) * HH;
#pragma unroll
    for (int v = 0; v < 8; ++v) {
        arow[8 * half + v]      = acc0[v] * inv;   // h = v + 8*half
        arow[16 + 8 * half + v] = acc1[v] * inv;   // h = 16 + v + 8*half
    }
    if (lane < 16) prop[(size_t)b * NN + ntile + lane] = accp[0] * inv;
}

// --- Kernel 3: combine + Wo projection (WMMA, transposed) + LayerNorm ---
__global__ void __launch_bounds__(32) k_epi(
    const float* __restrict__ feat, const float* __restrict__ att,
    const float* __restrict__ prop, const float* __restrict__ Wo,
    const float* __restrict__ bo, const float* __restrict__ gamma,
    const float* __restrict__ beta, float* __restrict__ out) {
    const int b  = blockIdx.x / (NN / 16);
    const int n0 = (blockIdx.x % (NN / 16)) * 16;
    const int lane = threadIdx.x, lm = lane & 15, half = lane >> 4;

    const float* frow = feat + ((size_t)b * NN + n0 + lm) * HH;
    const float* arow = att + ((size_t)b * NN + n0 + lm) * HH;
    const float  pr   = 0.1f * prop[(size_t)b * NN + n0 + lm];

    float c0[8], c1[8], a0[8], a1[8];
    *(f4*)&c0[0] = *(const f4*)(frow + 8 * half);      *(f4*)&c0[4] = *(const f4*)(frow + 8 * half + 4);
    *(f4*)&c1[0] = *(const f4*)(frow + 16 + 8 * half); *(f4*)&c1[4] = *(const f4*)(frow + 16 + 8 * half + 4);
    *(f4*)&a0[0] = *(const f4*)(arow + 8 * half);      *(f4*)&a0[4] = *(const f4*)(arow + 8 * half + 4);
    *(f4*)&a1[0] = *(const f4*)(arow + 16 + 8 * half); *(f4*)&a1[4] = *(const f4*)(arow + 16 + 8 * half + 4);
    v16bf bc;
#pragma unroll
    for (int v = 0; v < 8; ++v) {
        bc[v]     = (__bf16)(c0[v] + a0[v] + pr);
        bc[v + 8] = (__bf16)(c1[v] + a1[v] + pr);
    }
    const v16bf awo0 = cvt_row16(Wo + (size_t)lm * HH, half);
    const v16bf awo1 = cvt_row16(Wo + (size_t)(16 + lm) * HH, half);
    v8f z = {};
    v8f h0 = __builtin_amdgcn_wmma_f32_16x16x32_bf16(false, awo0, false, bc, (short)0, z, false, false);
    v8f h1 = __builtin_amdgcn_wmma_f32_16x16x32_bf16(false, awo1, false, bc, (short)0, z, false, false);

    float bo0[8], bo1[8];
    *(f4*)&bo0[0] = *(const f4*)(bo + 8 * half);      *(f4*)&bo0[4] = *(const f4*)(bo + 8 * half + 4);
    *(f4*)&bo1[0] = *(const f4*)(bo + 16 + 8 * half); *(f4*)&bo1[4] = *(const f4*)(bo + 16 + 8 * half + 4);
    float s = 0.f;
#pragma unroll
    for (int v = 0; v < 8; ++v) {
        h0[v] += bo0[v]; h1[v] += bo1[v];
        s += h0[v] + h1[v];
    }
    s += __shfl_xor(s, 16, 32);
    const float mu = s * (1.f / HH);
    float vs = 0.f;
#pragma unroll
    for (int v = 0; v < 8; ++v) {
        float d0 = h0[v] - mu, d1 = h1[v] - mu;
        vs += d0 * d0 + d1 * d1;
    }
    vs += __shfl_xor(vs, 16, 32);
    const float r = rsqrtf(vs * (1.f / HH) + LN_EPS);

    float g0[8], g1[8], bt0[8], bt1[8];
    *(f4*)&g0[0]  = *(const f4*)(gamma + 8 * half);      *(f4*)&g0[4]  = *(const f4*)(gamma + 8 * half + 4);
    *(f4*)&g1[0]  = *(const f4*)(gamma + 16 + 8 * half); *(f4*)&g1[4]  = *(const f4*)(gamma + 16 + 8 * half + 4);
    *(f4*)&bt0[0] = *(const f4*)(beta + 8 * half);       *(f4*)&bt0[4] = *(const f4*)(beta + 8 * half + 4);
    *(f4*)&bt1[0] = *(const f4*)(beta + 16 + 8 * half);  *(f4*)&bt1[4] = *(const f4*)(beta + 16 + 8 * half + 4);
    float o0[8], o1[8];
#pragma unroll
    for (int v = 0; v < 8; ++v) {
        o0[v] = (h0[v] - mu) * r * g0[v] + bt0[v];
        o1[v] = (h1[v] - mu) * r * g1[v] + bt1[v];
    }
    float* orow = out + ((size_t)b * NN + n0 + lm) * HH;  // n lane-fixed, h contiguous
    *(f4*)(orow + 8 * half)          = *(f4*)&o0[0];
    *(f4*)(orow + 8 * half + 4)      = *(f4*)&o0[4];
    *(f4*)(orow + 16 + 8 * half)     = *(f4*)&o1[0];
    *(f4*)(orow + 16 + 8 * half + 4) = *(f4*)&o1[4];
}

extern "C" void kernel_launch(void* const* d_in, const int* in_sizes, int n_in,
                              void* d_out, int out_size, void* d_ws, size_t ws_size,
                              hipStream_t stream) {
    const float* x     = (const float*)d_in[0];
    const float* feat  = (const float*)d_in[1];
    const float* dlog  = (const float*)d_in[2];
    const float* Wq    = (const float*)d_in[3];
    const float* bq    = (const float*)d_in[4];
    const float* Wk    = (const float*)d_in[5];
    const float* bk    = (const float*)d_in[6];
    const float* Wv    = (const float*)d_in[7];
    const float* bv    = (const float*)d_in[8];
    const float* adj   = (const float*)d_in[9];
    const float* geo   = (const float*)d_in[10];
    const float* Wo    = (const float*)d_in[11];
    const float* bo    = (const float*)d_in[12];
    const float* gamma = (const float*)d_in[13];
    const float* beta  = (const float*)d_in[14];
    float* out = (float*)d_out;

    char* ws = (char*)d_ws;
    size_t off = 0;
    __bf16* Qb = (__bf16*)(ws + off); off += (size_t)BB * NN * HH * 2;
    __bf16* Kb = (__bf16*)(ws + off); off += (size_t)BB * NN * HH * 2;
    __bf16* Vt = (__bf16*)(ws + off); off += (size_t)BB * NN * HH * 2;
    __bf16* Vd = (__bf16*)(ws + off); off += (size_t)BB * 16 * NN * 2;
    float* att  = (float*)(ws + off); off += (size_t)BB * NN * HH * 4;
    float* prop = (float*)(ws + off); off += (size_t)BB * NN * 4;
    (void)ws_size; (void)in_sizes; (void)n_in; (void)out_size;

    const int tiles = BB * (NN / 16);  // 2048 waves, one 16-row tile each
    k_delay_vd<<<(BB * NN + 255) / 256, 256, 0, stream>>>(x, dlog, Vd);
    k_qkvw<<<tiles, 32, 0, stream>>>(feat, Wq, bq, Wk, bk, Wv, bv, geo, Qb, Kb, Vt);
    k_flash<<<tiles, 32, 0, stream>>>(Qb, Kb, Vt, Vd, adj, geo, att, prop);
    k_epi<<<tiles, 32, 0, stream>>>(feat, att, prop, Wo, bo, gamma, beta, out);
}
